// AttEncode_66735201845257
// MI455X (gfx1250) — matrix operand order repaired and checked by reading, not compiled, not run on hardware
//
#include <hip/hip_runtime.h>
#include <cmath>

#define B_ 32
#define S_ 1024
#define D_ 200
#define H_ 8
#define STACK_ 6
#define KP_ 208          // D padded to 16-multiple (13 WMMA n-tiles)

typedef __attribute__((ext_vector_type(16))) _Float16 v16h;
typedef __attribute__((ext_vector_type(8)))  _Float16 v8h;
typedef __attribute__((ext_vector_type(8)))  float    v8f;

// ---------------------------------------------------------------------------
// WMMA GEMM on pre-packed f16 operands (no LDS, no guards in the hot loop).
//   A: f16 [z][M][Kp]   row-major, Kp % 32 == 0, pad cols are zero
//   B: f16 [z][Np][Kp]  column(n)-major, pad zero
// Outputs (either/both):
//   C : f32 [z][M][ldc]        (scale, +bias, +resid, relu fused)
//   Cp: f16 [z][M][ldcp]       packed copy, pad cols written as zero
// Block = 256 threads = 8 waves; wave computes 64(M) x 16(N) via 4 accums.
// ---------------------------------------------------------------------------
__global__ __launch_bounds__(256) void gemm_wmma_kernel(
    const _Float16* __restrict__ A, long long sA,
    const _Float16* __restrict__ B, long long sB,
    float*    __restrict__ C,  long long sC,  int ldc,
    _Float16* __restrict__ Cp, long long sCp, int ldcp,
    const float* __restrict__ bias,
    const float* __restrict__ resid, long long sR, int ldr,
    int M, int N, int Kp, float scale, int relu)
{
    const int lane = threadIdx.x & 31;
    const int wave = threadIdx.x >> 5;
    const int z    = blockIdx.z;
    const int m0   = blockIdx.x * 512 + wave * 64;   // wave owns 64 rows
    const int n0   = blockIdx.y * 16;

    const int g  = lane >> 4;     // half-wave group
    const int rl = lane & 15;     // M-row (A) / N-col (B,C) within tile

    const _Float16* __restrict__ Ab = A + (long long)z * sA;
    const _Float16* __restrict__ Bb = B + (long long)z * sB;

    const _Float16* __restrict__ ar0 = Ab + (long long)(m0 + rl) * Kp;
    const _Float16* __restrict__ ar1 = ar0 + (long long)16 * Kp;
    const _Float16* __restrict__ ar2 = ar0 + (long long)32 * Kp;
    const _Float16* __restrict__ ar3 = ar0 + (long long)48 * Kp;
    const _Float16* __restrict__ br  = Bb + (long long)(n0 + rl) * Kp;

    v8f acc0 = {}, acc1 = {}, acc2 = {}, acc3 = {};

    const int ka0 = 8 * g;        // A chunk 0: K = 8g .. 8g+7
    const int ka1 = 16 + 8 * g;   // A chunk 1: K = 16+8g .. 16+8g+7
    const int kb  = 8 * g;        // B chunks: K = 8g .. 8g+7, 16+8g .. 16+8g+7

    for (int k0 = 0; k0 < Kp; k0 += 32) {
        v8h b0 = *(const v8h*)(br + k0 + kb);
        v8h b1 = *(const v8h*)(br + k0 + kb + 16);
        v16h bf = __builtin_shufflevector(b0, b1,
                   0,1,2,3,4,5,6,7,8,9,10,11,12,13,14,15);

        v8h a00 = *(const v8h*)(ar0 + k0 + ka0);
        v8h a01 = *(const v8h*)(ar0 + k0 + ka1);
        v8h a10 = *(const v8h*)(ar1 + k0 + ka0);
        v8h a11 = *(const v8h*)(ar1 + k0 + ka1);
        v8h a20 = *(const v8h*)(ar2 + k0 + ka0);
        v8h a21 = *(const v8h*)(ar2 + k0 + ka1);
        v8h a30 = *(const v8h*)(ar3 + k0 + ka0);
        v8h a31 = *(const v8h*)(ar3 + k0 + ka1);

        v16h af0 = __builtin_shufflevector(a00, a01,
                    0,1,2,3,4,5,6,7,8,9,10,11,12,13,14,15);
        v16h af1 = __builtin_shufflevector(a10, a11,
                    0,1,2,3,4,5,6,7,8,9,10,11,12,13,14,15);
        v16h af2 = __builtin_shufflevector(a20, a21,
                    0,1,2,3,4,5,6,7,8,9,10,11,12,13,14,15);
        v16h af3 = __builtin_shufflevector(a30, a31,
                    0,1,2,3,4,5,6,7,8,9,10,11,12,13,14,15);

        acc0 = __builtin_amdgcn_wmma_f32_16x16x32_f16(false, af0, false, bf,
                                                      (short)0, acc0, false, false);
        acc1 = __builtin_amdgcn_wmma_f32_16x16x32_f16(false, af1, false, bf,
                                                      (short)0, acc1, false, false);
        acc2 = __builtin_amdgcn_wmma_f32_16x16x32_f16(false, af2, false, bf,
                                                      (short)0, acc2, false, false);
        acc3 = __builtin_amdgcn_wmma_f32_16x16x32_f16(false, af3, false, bf,
                                                      (short)0, acc3, false, false);
    }

    // fused epilogue
    const int  col = n0 + rl;
    const bool cok = col < N;
    float badd = 0.0f;
    if (cok && bias) badd = bias[col];

    v8f accs[4] = {acc0, acc1, acc2, acc3};
    #pragma unroll
    for (int mt = 0; mt < 4; ++mt) {
        #pragma unroll
        for (int r = 0; r < 8; ++r) {
            const int row = m0 + mt * 16 + g * 8 + r;
            float v = accs[mt][r] * scale + badd;
            if (cok && resid)
                v += resid[(long long)z * sR + (long long)row * ldr + col];
            if (relu) v = v > 0.0f ? v : 0.0f;
            if (C && cok)
                C[(long long)z * sC + (long long)row * ldc + col] = v;
            if (Cp)
                Cp[(long long)z * sCp + (long long)row * ldcp + col] =
                    cok ? (_Float16)v : (_Float16)0.0f;
        }
    }
}

// ---------------------------------------------------------------------------
// In-place f16 row softmax over S_=1024 cols. grid = B*S rows, 256 threads.
// ---------------------------------------------------------------------------
__global__ __launch_bounds__(256) void softmax_kernel(_Float16* __restrict__ p)
{
    __shared__ float red[256];
    _Float16* row = p + (long long)blockIdx.x * S_;
    const int t = threadIdx.x;

    float v0 = (float)row[t],       v1 = (float)row[t + 256];
    float v2 = (float)row[t + 512], v3 = (float)row[t + 768];

    float m = fmaxf(fmaxf(v0, v1), fmaxf(v2, v3));
    red[t] = m; __syncthreads();
    for (int s = 128; s > 0; s >>= 1) {
        if (t < s) red[t] = fmaxf(red[t], red[t + s]);
        __syncthreads();
    }
    m = red[0]; __syncthreads();

    v0 = __expf(v0 - m); v1 = __expf(v1 - m);
    v2 = __expf(v2 - m); v3 = __expf(v3 - m);

    red[t] = v0 + v1 + v2 + v3; __syncthreads();
    for (int s = 128; s > 0; s >>= 1) {
        if (t < s) red[t] += red[t + s];
        __syncthreads();
    }
    const float inv = 1.0f / red[0];

    row[t]       = (_Float16)(v0 * inv);
    row[t + 256] = (_Float16)(v1 * inv);
    row[t + 512] = (_Float16)(v2 * inv);
    row[t + 768] = (_Float16)(v3 * inv);
}

// ---------------------------------------------------------------------------
// LayerNorm over full (S,D) slab per batch; writes f32 h and packed f16 hP
// (incl. zeroing the pad columns 200..207). grid = B.
// ---------------------------------------------------------------------------
__global__ __launch_bounds__(256) void layernorm_kernel(
    const float* __restrict__ src, float* __restrict__ dst,
    _Float16* __restrict__ dstp)
{
    __shared__ float rs[256], rq[256];
    const long long base  = (long long)blockIdx.x * (long long)(S_ * D_);
    const long long basep = (long long)blockIdx.x * (long long)(S_ * KP_);
    const int t = threadIdx.x;

    float s = 0.0f, q = 0.0f;
    for (int i = t; i < S_ * D_; i += 256) {
        float v = src[base + i];
        s += v; q += v * v;
    }
    rs[t] = s; rq[t] = q; __syncthreads();
    for (int k = 128; k > 0; k >>= 1) {
        if (t < k) { rs[t] += rs[t + k]; rq[t] += rq[t + k]; }
        __syncthreads();
    }
    const float invN = 1.0f / (float)(S_ * D_);
    const float mean = rs[0] * invN;
    const float var  = rq[0] * invN - mean * mean;
    const float inv  = rsqrtf(var + 1e-5f);

    for (int i = t; i < S_ * D_; i += 256) {
        float v = (src[base + i] - mean) * inv;
        dst[base + i] = v;
        int sr = i / D_, d = i - sr * D_;
        dstp[basep + (long long)sr * KP_ + d] = (_Float16)v;
    }
    for (int i = t; i < S_ * (KP_ - D_); i += 256) {
        int sr = i >> 3, d = i & 7;
        dstp[basep + (long long)sr * KP_ + D_ + d] = (_Float16)0.0f;
    }
}

// ---------------------------------------------------------------------------
__global__ __launch_bounds__(256) void embed_kernel(
    const int* __restrict__ x, const float* __restrict__ emb,
    const float* __restrict__ pos, float* __restrict__ h,
    _Float16* __restrict__ hP)
{
    long long i = (long long)blockIdx.x * 256 + threadIdx.x;
    if (i >= (long long)B_ * S_ * D_) return;
    int d = (int)(i % D_);
    long long bs = i / D_;
    int s = (int)(bs % S_);
    float v = emb[(long long)x[bs] * D_ + d] + pos[s * D_ + d];
    h[i] = v;
    hP[bs * KP_ + d] = (_Float16)v;
}

__global__ __launch_bounds__(256) void zero_pad_rows_kernel(
    _Float16* __restrict__ p, long long rows)
{
    long long i = (long long)blockIdx.x * 256 + threadIdx.x;
    if (i >= rows * (KP_ - D_)) return;
    long long r = i >> 3; int d = (int)(i & 7);
    p[r * KP_ + D_ + d] = (_Float16)0.0f;
}

// Pack weight [K,N] f32 -> col-major f16 [208 n][208 k]; fold=1 sums the H
// identical head blocks of fuse_w [H*D, D].
__global__ __launch_bounds__(256) void pack_weight_kernel(
    const float* __restrict__ src, _Float16* __restrict__ dst, int fold)
{
    int i = blockIdx.x * 256 + threadIdx.x;
    if (i >= KP_ * KP_) return;
    int n = i / KP_, k = i - n * KP_;
    float v = 0.0f;
    if (n < D_ && k < D_) {
        if (fold) {
            #pragma unroll
            for (int hh = 0; hh < H_; ++hh)
                v += src[((long long)(hh * D_ + k)) * D_ + n];
        } else {
            v = src[(long long)k * D_ + n];
        }
    }
    dst[i] = (_Float16)v;
}

// qkvT[b][n][k] = qkvP[b][k][n]  (f16 transpose pack for attn@V B operand)
__global__ __launch_bounds__(256) void pack_transpose_kernel(
    const _Float16* __restrict__ src, _Float16* __restrict__ dst)
{
    long long i = (long long)blockIdx.x * 256 + threadIdx.x;
    if (i >= (long long)B_ * KP_ * S_) return;
    int k = (int)(i % S_);
    long long bn = i / S_;
    int n = (int)(bn % KP_);
    long long b = bn / KP_;
    dst[i] = src[b * (long long)S_ * KP_ + (long long)k * KP_ + n];
}

__global__ __launch_bounds__(256) void copy_kernel(
    const float* __restrict__ src, float* __restrict__ dst, long long n)
{
    long long i = (long long)blockIdx.x * 256 + threadIdx.x;
    if (i < n) dst[i] = src[i];
}

// ---------------------------------------------------------------------------
extern "C" void kernel_launch(void* const* d_in, const int* in_sizes, int n_in,
                              void* d_out, int out_size, void* d_ws, size_t ws_size,
                              hipStream_t stream)
{
    const int*   x      = (const int*)  d_in[0];
    const float* embedw = (const float*)d_in[1];
    const float* pos    = (const float*)d_in[2];
    const float* qkv_w  = (const float*)d_in[3];
    const float* qkv_b  = (const float*)d_in[4];
    const float* fuse_w = (const float*)d_in[5];
    const float* fuse_b = (const float*)d_in[6];
    const float* w1     = (const float*)d_in[7];
    const float* b1     = (const float*)d_in[8];
    const float* w2     = (const float*)d_in[9];
    const float* b2     = (const float*)d_in[10];

    const long long nBSD  = (long long)B_ * S_ * D_;    // 6.55M f32
    const long long nRows = (long long)B_ * S_;         // 32768
    const long long nPack = nRows * KP_;                // 6.82M f16
    const long long nBSS  = (long long)B_ * S_ * S_;    // 33.55M f16
    const long long nT    = (long long)B_ * KP_ * S_;   // 6.82M f16

    float* h = (float*)d_ws;                 // f32 activations
    float* y = h + nBSD;                     // f32 pre-LN buffer
    _Float16* hP      = (_Float16*)(y + nBSD);
    _Float16* qkvP    = hP + nPack;          // packed q=k=v [B,S,208]
    _Float16* ctxP    = qkvP + nPack;        // packed attn output
    _Float16* tP      = ctxP + nPack;        // packed FFN hidden
    _Float16* qkvT    = tP + nPack;          // [B,208,1024] transpose pack
    _Float16* scoresP = qkvT + nT;           // [B,S,S] f16 scores/probs
    _Float16* qkvwP   = scoresP + nBSS;      // packed weights [208,208]
    _Float16* fweffP  = qkvwP + KP_ * KP_;
    _Float16* w1P     = fweffP + KP_ * KP_;
    _Float16* w2P     = w1P + KP_ * KP_;

    const int WGRID = (KP_ * KP_ + 255) / 256;
    pack_weight_kernel<<<dim3(WGRID), 256, 0, stream>>>(qkv_w,  qkvwP,  0);
    pack_weight_kernel<<<dim3(WGRID), 256, 0, stream>>>(fuse_w, fweffP, 1);
    pack_weight_kernel<<<dim3(WGRID), 256, 0, stream>>>(w1,     w1P,    0);
    pack_weight_kernel<<<dim3(WGRID), 256, 0, stream>>>(w2,     w2P,    0);

    embed_kernel<<<dim3((unsigned)((nBSD + 255) / 256)), 256, 0, stream>>>(
        x, embedw, pos, h, hP);
    zero_pad_rows_kernel<<<dim3((unsigned)((nRows * 8 + 255) / 256)), 256, 0,
                           stream>>>(hP, nRows);

    const int M  = B_ * S_;           // 32768
    const int NT = KP_ / 16;          // 13 n-tiles
    const float inv_sqrt_d = 1.0f / sqrtf((float)D_);
    const long long sPk = (long long)S_ * KP_;   // per-batch packed stride
    const long long sSS = (long long)S_ * S_;

    for (int layer = 0; layer < STACK_; ++layer) {
        // qkvP = pack16( h @ qkv_w + qkv_b )
        gemm_wmma_kernel<<<dim3(M / 512, NT, 1), 256, 0, stream>>>(
            hP, 0, qkvwP, 0,
            nullptr, 0, 0, qkvP, 0, KP_,
            qkv_b, nullptr, 0, 0, M, D_, KP_, 1.0f, 0);

        // scoresP = pack16( (qkv @ qkv^T) / sqrt(D) )   per batch
        gemm_wmma_kernel<<<dim3(S_ / 512, S_ / 16, B_), 256, 0, stream>>>(
            qkvP, sPk, qkvP, sPk,
            nullptr, 0, 0, scoresP, sSS, S_,
            nullptr, nullptr, 0, 0, S_, S_, KP_, inv_sqrt_d, 0);

        softmax_kernel<<<dim3(B_ * S_), 256, 0, stream>>>(scoresP);

        // qkvT = transpose pack of qkvP for the B operand
        pack_transpose_kernel<<<dim3((unsigned)((nT + 255) / 256)), 256, 0,
                                stream>>>(qkvP, qkvT);

        // ctxP = pack16( probs @ qkv )   per batch, K = 1024
        gemm_wmma_kernel<<<dim3(S_ / 512, NT, B_), 256, 0, stream>>>(
            scoresP, sSS, qkvT, (long long)KP_ * S_,
            nullptr, 0, 0, ctxP, sPk, KP_,
            nullptr, nullptr, 0, 0, S_, D_, S_, 1.0f, 0);

        // y = ctx @ fweff + fuse_b + h   (head concat folded into fweff)
        gemm_wmma_kernel<<<dim3(M / 512, NT, 1), 256, 0, stream>>>(
            ctxP, 0, fweffP, 0,
            y, 0, D_, nullptr, 0, 0,
            fuse_b, h, 0, D_, M, D_, KP_, 1.0f, 0);
        layernorm_kernel<<<dim3(B_), 256, 0, stream>>>(y, h, hP);

        // FFN: tP = pack16(relu(h@w1+b1)); y = t@w2 + b2 + h; h = LN(y)
        gemm_wmma_kernel<<<dim3(M / 512, NT, 1), 256, 0, stream>>>(
            hP, 0, w1P, 0,
            nullptr, 0, 0, tP, 0, KP_,
            b1, nullptr, 0, 0, M, D_, KP_, 1.0f, 1);
        gemm_wmma_kernel<<<dim3(M / 512, NT, 1), 256, 0, stream>>>(
            tP, 0, w2P, 0,
            y, 0, D_, nullptr, 0, 0,
            b2, h, 0, D_, M, D_, KP_, 1.0f, 0);
        layernorm_kernel<<<dim3(B_), 256, 0, stream>>>(y, h, hP);
    }

    copy_kernel<<<dim3((unsigned)((nBSD + 255) / 256)), 256, 0, stream>>>(
        h, (float*)d_out, nBSD);
}
